// NodeEncoder_28080496181844
// MI455X (gfx1250) — compile-verified, matrix operand order checked by reading, not run on hardware
//
#include <hip/hip_runtime.h>
#include <hip/hip_bf16.h>

typedef __bf16 bf16_t;
typedef __attribute__((ext_vector_type(16))) __bf16 v16bf;
typedef __attribute__((ext_vector_type(8)))  float  v8f;

constexpr int N_NODES = 50000;
constexpr int N_EDGES = 400000;
constexpr int NODES_PER_GRAPH = 100;

// ---------------------------------------------------------------- WMMA helpers

__device__ inline v8f wmma_bf16(v16bf a, v16bf b, v8f c) {
  // D = A(16x32) * B(32x16) + C, f32 accumulate
  return __builtin_amdgcn_wmma_f32_16x16x32_bf16(false, a, false, b, (short)0, c,
                                                 false, false);
}

// Load a 16x32 bf16 fragment from a row-major tile (row stride in elements).
// Per CDNA5 ISA 16-bit A layout: lane<16 -> row=lane, K = {0..7} u {16..23};
// lane>=16 -> row=lane-16, K = {8..15} u {24..31}. B uses the mirrored layout
// with N as the "row", so a [N][K] (transposed-weight) tile loads identically.
__device__ inline v16bf load_frag(const bf16_t* __restrict__ base, int stride) {
  const int lane = threadIdx.x & 31;
  const bf16_t* p = base + (lane & 15) * stride + ((lane >> 4) << 3);
  union { uint4 q[2]; v16bf v; } u;
  u.q[0] = *(const uint4*)(p);
  u.q[1] = *(const uint4*)(p + 16);
  return u.v;
}

__device__ inline float wave_reduce_sum(float v) {
  #pragma unroll
  for (int m = 16; m >= 1; m >>= 1) v += __shfl_xor(v, m, 32);
  return v;
}

__device__ inline void atomic_max_f32(float* a, float v) {
  int iv = __float_as_int(v);
  if (iv >= 0) atomicMax((int*)a, iv);
  else         atomicMin((unsigned int*)a, (unsigned int)iv);
}

// ---------------------------------------------------------------- small kernels

// Wt[n*K + k] = bf16(W[k*N + n])   (W is [K][N] row-major fp32)
__global__ void k_transpose_bf16(const float* __restrict__ W, bf16_t* __restrict__ Wt,
                                 int K, int N) {
  int i = blockIdx.x * blockDim.x + threadIdx.x;
  if (i >= K * N) return;
  int n = i / K, k = i - n * K;
  Wt[i] = (bf16_t)W[k * N + n];
}

__global__ void k_init(float* __restrict__ sent, float* __restrict__ recv,
                       float* __restrict__ m_s, float* __restrict__ m_r,
                       float* __restrict__ z_s, float* __restrict__ z_r) {
  int i = blockIdx.x * blockDim.x + threadIdx.x;
  if (i < N_NODES * 128) { sent[i] = 0.f; recv[i] = 0.f; }
  if (i < N_NODES * 2) {
    z_s[i] = 0.f; z_r[i] = 0.f;
    float ninf = __int_as_float(0xff800000u);
    m_s[i] = ninf; m_r[i] = ninf;
  }
}

__global__ void k_segmax(const float* __restrict__ L, const int* __restrict__ idx,
                         float* __restrict__ m) {
  int i = blockIdx.x * blockDim.x + threadIdx.x;
  if (i >= N_EDGES * 2) return;
  atomic_max_f32(&m[idx[i >> 1] * 2 + (i & 1)], L[i]);
}

__global__ void k_expsum(const float* __restrict__ L, const int* __restrict__ idx,
                         const float* __restrict__ m, float* __restrict__ expv,
                         float* __restrict__ z) {
  int i = blockIdx.x * blockDim.x + threadIdx.x;
  if (i >= N_EDGES * 2) return;
  int node = idx[i >> 1], h = i & 1;
  float ev = __expf(L[i] - m[node * 2 + h]);
  expv[i] = ev;
  atomicAdd(&z[node * 2 + h], ev);
}

__global__ void k_scatter(const float* __restrict__ expv, const float* __restrict__ z,
                          const int* __restrict__ idx, const bf16_t* __restrict__ v,
                          float* __restrict__ out) {
  int i = blockIdx.x * blockDim.x + threadIdx.x;  // over E*128
  int e = i >> 7, d = i & 127, h = d >> 6;
  int node = idx[e];
  float alpha = expv[e * 2 + h] / (z[node * 2 + h] + 1e-16f);
  atomicAdd(&out[node * 128 + d], alpha * (float)v[i]);
}

// ---------------------------------------------------------------- edge kernel
// One block = 16 edges. Computes both attn calls (shared LDS concat tile):
//   h = leaky_relu([.|.|ea] @ w1 + b1), logits = <h, w2>, v = [.|ea] @ w3 + b3

__global__ __launch_bounds__(256)
void k_edge(const int* __restrict__ edges, const float* __restrict__ feat,
            const float* __restrict__ ea, const float* __restrict__ b1,
            const float* __restrict__ w2, const float* __restrict__ b3,
            const bf16_t* __restrict__ w1t, const bf16_t* __restrict__ w3t,
            float* __restrict__ logits_s, float* __restrict__ logits_r,
            bf16_t* __restrict__ v_s, bf16_t* __restrict__ v_r) {
  constexpr int XS = 3 * 128 + 8;  // bf16 row stride (bank rotate)
  constexpr int HS = 128 + 4;      // f32 row stride
  __shared__ bf16_t xcat[16 * XS];   // [Q(0:128) | K(128:256) | E(256:384)]
  __shared__ float hs[16 * HS];
  __shared__ float hr[16 * HS];
  __shared__ int rowi[16], coli[16];

  const int tid = threadIdx.x, lane = tid & 31, wave = tid >> 5;
  const int e0 = blockIdx.x * 16;

  if (tid < 16) { rowi[tid] = edges[e0 + tid]; coli[tid] = edges[N_EDGES + e0 + tid]; }
  __syncthreads();

  for (int i = tid; i < 16 * 128; i += 256) {
    int e = i >> 7, c = i & 127;
    xcat[e * XS + c]       = (bf16_t)feat[(size_t)rowi[e] * 128 + c];
    xcat[e * XS + 128 + c] = (bf16_t)feat[(size_t)coli[e] * 128 + c];
    xcat[e * XS + 256 + c] = (bf16_t)ea[(size_t)(e0 + e) * 128 + c];
  }
  __syncthreads();

  const int n0 = wave * 16;  // each wave owns one 16-wide N tile

  // ---- h GEMMs: sent order [Q,K,E], recv order [K,Q,E]; shared B fragments
  {
    v8f acc_s = {}, acc_r = {};
    #pragma unroll
    for (int kc = 0; kc < 12; ++kc) {
      int offS = kc * 32;
      int offR = (kc < 4) ? 128 + kc * 32 : (kc < 8 ? (kc - 4) * 32 : 256 + (kc - 8) * 32);
      v16bf b = load_frag(w1t + n0 * 384 + kc * 32, 384);
      acc_s = wmma_bf16(load_frag(xcat + offS, XS), b, acc_s);
      acc_r = wmma_bf16(load_frag(xcat + offR, XS), b, acc_r);
    }
    int n = n0 + (lane & 15), mb = (lane >> 4) * 8;
    float bv = b1[n];
    #pragma unroll
    for (int v = 0; v < 8; ++v) {
      float s = acc_s[v] + bv; hs[(mb + v) * HS + n] = s > 0.f ? s : 0.2f * s;
      float r = acc_r[v] + bv; hr[(mb + v) * HS + n] = r > 0.f ? r : 0.2f * r;
    }
  }
  __syncthreads();

  // ---- logits: 16 edges x 2 heads, one lane each (waves 0/1)
  if (wave < 2) {
    const float* H = wave ? hr : hs;
    float* L = wave ? logits_r : logits_s;
    int e = lane >> 1, h = lane & 1;
    float acc = 0.f;
    #pragma unroll
    for (int d = 0; d < 64; ++d) acc += H[e * HS + h * 64 + d] * w2[h * 64 + d];
    L[(e0 + e) * 2 + h] = acc;
  }

  // ---- v GEMMs: sent=[K,E] (contiguous 128..383), recv=[Q,E]
  {
    v8f acc_s = {}, acc_r = {};
    #pragma unroll
    for (int kc = 0; kc < 8; ++kc) {
      int offS = 128 + kc * 32;
      int offR = (kc < 4) ? kc * 32 : 256 + (kc - 4) * 32;
      v16bf b = load_frag(w3t + n0 * 256 + kc * 32, 256);
      acc_s = wmma_bf16(load_frag(xcat + offS, XS), b, acc_s);
      acc_r = wmma_bf16(load_frag(xcat + offR, XS), b, acc_r);
    }
    int n = n0 + (lane & 15), mb = (lane >> 4) * 8;
    float bv = b3[n];
    #pragma unroll
    for (int v = 0; v < 8; ++v) {
      v_s[(size_t)(e0 + mb + v) * 128 + n] = (bf16_t)(acc_s[v] + bv);
      v_r[(size_t)(e0 + mb + v) * 128 + n] = (bf16_t)(acc_r[v] + bv);
    }
  }
}

// ---------------------------------------------------------------- node MLP
constexpr int MXS = 512 + 8;  // bf16 stride
constexpr int MHS = 512 + 4;  // f32 stride

__device__ inline void gemm512(const bf16_t* __restrict__ X,
                               const bf16_t* __restrict__ Wt,
                               const float* __restrict__ bias,
                               float* __restrict__ H, int wave) {
  const int lane = threadIdx.x & 31;
  #pragma unroll
  for (int t = 0; t < 4; ++t) {
    int n0 = (wave + t * 8) * 16;
    v8f acc = {};
    #pragma unroll
    for (int kc = 0; kc < 16; ++kc)
      acc = wmma_bf16(load_frag(X + kc * 32, MXS),
                      load_frag(Wt + n0 * 512 + kc * 32, 512), acc);
    int n = n0 + (lane & 15), mb = (lane >> 4) * 8;
    float bv = bias[n];
    #pragma unroll
    for (int v = 0; v < 8; ++v) H[(mb + v) * MHS + n] = acc[v] + bv;
  }
}

__device__ inline void ln_relu(const float* __restrict__ H, bf16_t* __restrict__ O,
                               const float* __restrict__ sc, const float* __restrict__ bi,
                               int wave) {
  const int lane = threadIdx.x & 31;
  #pragma unroll
  for (int rr = 0; rr < 2; ++rr) {
    int r = wave * 2 + rr;
    float s = 0.f;
    for (int c = lane; c < 512; c += 32) s += H[r * MHS + c];
    float mu = wave_reduce_sum(s) * (1.f / 512.f);
    float vs = 0.f;
    for (int c = lane; c < 512; c += 32) { float d = H[r * MHS + c] - mu; vs += d * d; }
    float rstd = __frsqrt_rn(wave_reduce_sum(vs) * (1.f / 512.f) + 1e-5f);
    for (int c = lane; c < 512; c += 32) {
      float y = (H[r * MHS + c] - mu) * rstd * sc[c] + bi[c];
      O[r * MXS + c] = (bf16_t)fmaxf(y, 0.f);
    }
  }
}

__global__ __launch_bounds__(256)
void k_mlp(const float* __restrict__ feat, const float* __restrict__ sent,
           const float* __restrict__ recv, const float* __restrict__ u,
           const bf16_t* __restrict__ mw0t, const float* __restrict__ mb0,
           const float* __restrict__ ls0, const float* __restrict__ lb0,
           const bf16_t* __restrict__ mw1t, const float* __restrict__ mb1,
           const float* __restrict__ ls1, const float* __restrict__ lb1,
           const bf16_t* __restrict__ mw2t, const float* __restrict__ mb2,
           float* __restrict__ out) {
  __shared__ bf16_t xb[16 * MXS];  // reused as LN output buffer
  __shared__ float  hf[16 * MHS];
  const int tid = threadIdx.x, lane = tid & 31, wave = tid >> 5;
  const int node0 = blockIdx.x * 16;

  for (int i = tid; i < 16 * 128; i += 256) {
    int m = i >> 7, c = i & 127;
    int node = node0 + m;
    xb[m * MXS + c]       = (bf16_t)feat[(size_t)node * 128 + c];
    xb[m * MXS + 128 + c] = (bf16_t)sent[(size_t)node * 128 + c];
    xb[m * MXS + 256 + c] = (bf16_t)recv[(size_t)node * 128 + c];
    xb[m * MXS + 384 + c] = (bf16_t)u[(size_t)(node / NODES_PER_GRAPH) * 128 + c];
  }
  __syncthreads();

  gemm512(xb, mw0t, mb0, hf, wave);   __syncthreads();
  ln_relu(hf, xb, ls0, lb0, wave);    __syncthreads();
  gemm512(xb, mw1t, mb1, hf, wave);   __syncthreads();
  ln_relu(hf, xb, ls1, lb1, wave);    __syncthreads();

  // layer2: 512 -> 128, one N-tile per wave
  {
    int n0 = wave * 16;
    v8f acc = {};
    #pragma unroll
    for (int kc = 0; kc < 16; ++kc)
      acc = wmma_bf16(load_frag(xb + kc * 32, MXS),
                      load_frag(mw2t + n0 * 512 + kc * 32, 512), acc);
    int n = n0 + (lane & 15), mb = (lane >> 4) * 8;
    float bv = mb2[n];
    #pragma unroll
    for (int v = 0; v < 8; ++v)
      out[(size_t)(node0 + mb + v) * 128 + n] = acc[v] + bv;
  }
}

// ---------------------------------------------------------------- launch

extern "C" void kernel_launch(void* const* d_in, const int* in_sizes, int n_in,
                              void* d_out, int out_size, void* d_ws, size_t ws_size,
                              hipStream_t stream) {
  (void)in_sizes; (void)n_in; (void)out_size; (void)ws_size;
  const int*   edges = (const int*)d_in[0];
  const float* feat  = (const float*)d_in[1];
  const float* eattr = (const float*)d_in[2];
  const float* u     = (const float*)d_in[3];
  const float* w1  = (const float*)d_in[5];
  const float* b1  = (const float*)d_in[6];
  const float* w2  = (const float*)d_in[7];
  const float* w3  = (const float*)d_in[8];
  const float* b3  = (const float*)d_in[9];
  const float* mw0 = (const float*)d_in[10];
  const float* mb0 = (const float*)d_in[11];
  const float* ls0 = (const float*)d_in[12];
  const float* lb0 = (const float*)d_in[13];
  const float* mw1 = (const float*)d_in[14];
  const float* mb1 = (const float*)d_in[15];
  const float* ls1 = (const float*)d_in[16];
  const float* lb1 = (const float*)d_in[17];
  const float* mw2 = (const float*)d_in[18];
  const float* mb2 = (const float*)d_in[19];
  float* out = (float*)d_out;

  char* ws = (char*)d_ws;
  size_t off = 0;
  auto alloc = [&](size_t bytes) {
    void* p = ws + off; off += (bytes + 255) & ~(size_t)255; return p;
  };
  bf16_t* w1t  = (bf16_t*)alloc((size_t)128 * 384 * 2);
  bf16_t* w3t  = (bf16_t*)alloc((size_t)128 * 256 * 2);
  bf16_t* mw0t = (bf16_t*)alloc((size_t)512 * 512 * 2);
  bf16_t* mw1t = (bf16_t*)alloc((size_t)512 * 512 * 2);
  bf16_t* mw2t = (bf16_t*)alloc((size_t)128 * 512 * 2);
  float* logits_s = (float*)alloc((size_t)N_EDGES * 2 * 4);
  float* logits_r = (float*)alloc((size_t)N_EDGES * 2 * 4);
  float* exp_s = (float*)alloc((size_t)N_EDGES * 2 * 4);
  float* exp_r = (float*)alloc((size_t)N_EDGES * 2 * 4);
  float* m_s = (float*)alloc((size_t)N_NODES * 2 * 4);
  float* m_r = (float*)alloc((size_t)N_NODES * 2 * 4);
  float* z_s = (float*)alloc((size_t)N_NODES * 2 * 4);
  float* z_r = (float*)alloc((size_t)N_NODES * 2 * 4);
  float* sent = (float*)alloc((size_t)N_NODES * 128 * 4);
  float* recv = (float*)alloc((size_t)N_NODES * 128 * 4);
  bf16_t* v_s = (bf16_t*)alloc((size_t)N_EDGES * 128 * 2);
  bf16_t* v_r = (bf16_t*)alloc((size_t)N_EDGES * 128 * 2);

  const int T = 256;
  k_transpose_bf16<<<(384 * 128 + T - 1) / T, T, 0, stream>>>(w1, w1t, 384, 128);
  k_transpose_bf16<<<(256 * 128 + T - 1) / T, T, 0, stream>>>(w3, w3t, 256, 128);
  k_transpose_bf16<<<(512 * 512 + T - 1) / T, T, 0, stream>>>(mw0, mw0t, 512, 512);
  k_transpose_bf16<<<(512 * 512 + T - 1) / T, T, 0, stream>>>(mw1, mw1t, 512, 512);
  k_transpose_bf16<<<(512 * 128 + T - 1) / T, T, 0, stream>>>(mw2, mw2t, 512, 128);
  k_init<<<(N_NODES * 128 + T - 1) / T, T, 0, stream>>>(sent, recv, m_s, m_r, z_s, z_r);

  k_edge<<<N_EDGES / 16, 256, 0, stream>>>(edges, feat, eattr, b1, w2, b3,
                                           w1t, w3t, logits_s, logits_r, v_s, v_r);

  const int* row = edges;
  const int* col = edges + N_EDGES;
  k_segmax<<<(N_EDGES * 2 + T - 1) / T, T, 0, stream>>>(logits_s, row, m_s);
  k_segmax<<<(N_EDGES * 2 + T - 1) / T, T, 0, stream>>>(logits_r, col, m_r);
  k_expsum<<<(N_EDGES * 2 + T - 1) / T, T, 0, stream>>>(logits_s, row, m_s, exp_s, z_s);
  k_expsum<<<(N_EDGES * 2 + T - 1) / T, T, 0, stream>>>(logits_r, col, m_r, exp_r, z_r);
  k_scatter<<<N_EDGES * 128 / T, T, 0, stream>>>(exp_s, z_s, row, v_s, sent);
  k_scatter<<<N_EDGES * 128 / T, T, 0, stream>>>(exp_r, z_r, col, v_r, recv);

  k_mlp<<<N_NODES / 16, 256, 0, stream>>>(feat, sent, recv, u,
                                          mw0t, mb0, ls0, lb0,
                                          mw1t, mb1, ls1, lb1,
                                          mw2t, mb2, out);
}
